// RenderingNetwork_87754771792672
// MI455X (gfx1250) — matrix operand order repaired and checked by reading, not compiled, not run on hardware
//
#include <hip/hip_runtime.h>
#include <math.h>

// ---------------- problem constants (from reference) ----------------
#define RR    512
#define SS    48
#define NPTS  (RR * SS)      // 24576
#define MPART 4096
#define KNN   16
#define RADIUS 0.1f
#define D_PAD 512            // 511 features padded to 512 for K%32==0
#define D_HID 256

// ---------------- CDNA5 WMMA types ----------------
typedef __attribute__((ext_vector_type(16))) __bf16 v16bf;
typedef __attribute__((ext_vector_type(8)))  float  v8f;

union FragU { uint4 u4[2]; v16bf v; };

// ---------------- bf16 helpers (round-to-nearest-even) ----------------
__device__ __forceinline__ unsigned short f2bf(float f) {
  unsigned u = __float_as_uint(f);
  u += 0x7FFFu + ((u >> 16) & 1u);
  return (unsigned short)(u >> 16);
}
__device__ __forceinline__ float bf2f(unsigned short h) {
  return __uint_as_float(((unsigned)h) << 16);
}

// ---------------- posenc writer: [x, sin(2^0 x), cos(2^0 x), ...] ----------------
__device__ __forceinline__ int emit_posenc(unsigned short* row, int off,
                                           const float* x, int c, int nf) {
  for (int j = 0; j < c; ++j) row[off + j] = f2bf(x[j]);
  off += c;
  float s = 1.f;
  for (int f = 0; f < nf; ++f) {
    for (int j = 0; j < c; ++j) row[off + j]     = f2bf(__sinf(x[j] * s));
    for (int j = 0; j < c; ++j) row[off + c + j] = f2bf(__cosf(x[j] * s));
    off += 2 * c;
    s *= 2.f;
  }
  return off;
}

// ---------------- weight conversion: f32 -> bf16 (W0 padded 511->512) ----------------
__global__ void __launch_bounds__(256)
convert_weights(const float* __restrict__ W0, const float* __restrict__ W1,
                const float* __restrict__ W2,
                unsigned short* __restrict__ Wb0, unsigned short* __restrict__ Wb1,
                unsigned short* __restrict__ Wb2) {
  int idx = blockIdx.x * 256 + threadIdx.x;            // 262144 total
  if (idx < 256 * 512) {
    int n = idx >> 9, k = idx & 511;
    Wb0[idx] = f2bf(k < 511 ? W0[n * 511 + k] : 0.f);
  } else if (idx < 256 * 512 + 256 * 256) {
    int j = idx - 256 * 512;
    Wb1[j] = f2bf(W1[j]);
  } else {
    int j = idx - 256 * 512 - 256 * 256;
    Wb2[j] = f2bf(W2[j]);
  }
}

// ---------------- ball query + feature assembly (one thread per point) ----------------
__global__ void __launch_bounds__(256)
build_features(const float* __restrict__ points,
               const float* __restrict__ normals,
               const float* __restrict__ feats,
               const float* __restrict__ particles,
               const float* __restrict__ rays,
               const float* __restrict__ ro,
               unsigned short* __restrict__ X0) {
  __shared__ float sP[MPART * 3];                      // 48 KB << 320 KB WGP LDS
  const int tid = threadIdx.x;
  for (int j = tid; j < MPART * 3; j += 256) sP[j] = particles[j];
  __syncthreads();

  const int i = blockIdx.x * 256 + tid;                // exact: NPTS/256 blocks
  const float qx = points[i * 3 + 0], qy = points[i * 3 + 1], qz = points[i * 3 + 2];
  const float R2 = RADIUS * RADIUS;

  int cnt = 0, nvar = 0;
  float density = 0.f, smx = 0.f, smy = 0.f, smz = 0.f;
  float s1x = 0.f, s1y = 0.f, s1z = 0.f, s2x = 0.f, s2y = 0.f, s2z = 0.f;

  for (int j = 0; j < MPART; ++j) {
    float px = sP[3 * j + 0], py = sP[3 * j + 1], pz = sP[3 * j + 2];
    float dx = qx - px, dy = qy - py, dz = qz - pz;
    float d2 = dx * dx + dy * dy + dz * dz;
    if (d2 < R2 && cnt < KNN) {                        // first-K by index order
      ++cnt;
      float t = sqrtf(d2) * (1.f / RADIUS);
      float w = fmaxf(1.f - t * t * t, 0.f);
      density += w;
      smx += w * px; smy += w * py; smz += w * pz;
      if (d2 != 0.f) {                                 // nn_mask = dist != 0
        ++nvar;
        s1x -= dx; s1y -= dy; s1z -= dz;               // diff = nbr - q = -d
        s2x += dx * dx; s2y += dy * dy; s2z += dz * dz;
      }
    }
    if ((j & 511) == 511) { if (__ballot(cnt < KNN) == 0ull) break; }
  }
  // padded slots contribute nbr=(0,0,0) to smoothing (dn = ||q||)
  {
    float t0 = sqrtf(qx * qx + qy * qy + qz * qz) * (1.f / RADIUS);
    float w0 = fmaxf(1.f - t0 * t0 * t0, 0.f);
    density += (float)(KNN - cnt) * w0;
  }
  const float inv_nn = 1.f / ((float)nvar + 1e-12f);
  float mx = s1x * inv_nn, my = s1y * inv_nn, mz = s1z * inv_nn;
  float vx = fmaxf(s2x * inv_nn - mx * mx, 0.f);
  float vy = fmaxf(s2y * inv_nn - my * my, 0.f);
  float vz = fmaxf(s2z * inv_nn - mz * mz, 0.f);

  const float invd = 1.f / (density + 1e-12f);
  float spx = smx * invd, spy = smy * invd, spz = smz * invd;
  float dxr = spx - ro[0], dyr = spy - ro[1], dzr = spz - ro[2];
  float rl = rsqrtf(dxr * dxr + dyr * dyr + dzr * dzr);
  dxr *= rl; dyr *= rl; dzr *= rl;

  const int r = i / SS;
  float rd[3]  = { rays[r * 6 + 3], rays[r * 6 + 4], rays[r * 6 + 5] };
  float q3[3]  = { qx, qy, qz };
  float sp3[3] = { spx, spy, spz };
  float v3[3]  = { vx, vy, vz };
  float dr3[3] = { dxr, dyr, dzr };
  float den1[1] = { density };

  unsigned short* row = X0 + (size_t)i * D_PAD;
  int off = 0;
  off = emit_posenc(row, off, q3,  3, 10);   // hit_pos_emb      -> 63
  off = emit_posenc(row, off, den1, 1, 4);   // density_emb      -> 72
  off = emit_posenc(row, off, sp3, 3, 10);   // smoothed_pos_emb -> 135
  off = emit_posenc(row, off, v3,  3, 10);   // var_emb          -> 198
  off = emit_posenc(row, off, rd,  3, 4);    // hit_dir_emb      -> 225
  off = emit_posenc(row, off, dr3, 3, 4);    // posenc(dirs,4)   -> 252
  row[off++] = f2bf(normals[i * 3 + 0]);
  row[off++] = f2bf(normals[i * 3 + 1]);
  row[off++] = f2bf(normals[i * 3 + 2]);     // -> 255
  const float* fv = feats + (size_t)i * 256;
  for (int k = 0; k < 256; ++k) row[off + k] = f2bf(fv[k]);  // -> 511
  row[511] = 0;                                              // pad column
}

// ---------------- bf16 WMMA GEMM: Y = relu(X @ W^T + b), bf16 out ----------------
// X: [Mrows,K] bf16, W: [N,K] bf16 row-major, block = 64(M) x 256(N), 8 waves (2Mx4N)
__global__ void __launch_bounds__(256)
gemm_bf16_relu(const unsigned short* __restrict__ X,
               const unsigned short* __restrict__ W,
               const float* __restrict__ bias,
               unsigned short* __restrict__ Y,
               int K, int N) {
  const int tid   = threadIdx.x;
  const int wave  = tid >> 5, lane = tid & 31;
  const int waveM = wave & 1, waveN = wave >> 1;
  const int mBase = blockIdx.x * 64 + waveM * 32;
  const int nBase = waveN * 64;
  const int lm = lane & 15, kh = lane >> 4;

  // Incrementing fragment pointers (64 B per K-step); chunk offsets are constant:
  //   A (16x32 bf16): lane holds row lm; chunks [kh*8,+8) and [16+kh*8,+8) -> p[0], p[2]
  //   B (32x16 bf16): lane holds col lm; chunks [kh*8,+8) and [16+kh*8,+8) -> p[0], p[2]
  const uint4* pA0 = (const uint4*)(X + (size_t)(mBase +      lm) * K) + kh;
  const uint4* pA1 = (const uint4*)(X + (size_t)(mBase + 16 + lm) * K) + kh;
  const uint4* pB0 = (const uint4*)(W + (size_t)(nBase +  0 + lm) * K) + kh;
  const uint4* pB1 = (const uint4*)(W + (size_t)(nBase + 16 + lm) * K) + kh;
  const uint4* pB2 = (const uint4*)(W + (size_t)(nBase + 32 + lm) * K) + kh;
  const uint4* pB3 = (const uint4*)(W + (size_t)(nBase + 48 + lm) * K) + kh;

  v8f acc[2][4];
#pragma unroll
  for (int t = 0; t < 2; ++t)
#pragma unroll
    for (int u = 0; u < 4; ++u)
#pragma unroll
      for (int j = 0; j < 8; ++j) acc[t][u][j] = 0.f;

#pragma unroll 2
  for (int k0 = 0; k0 < K; k0 += 32) {
    // Speculative prefetch (gfx1250 global_prefetch_b8): safe past end of row,
    // translation failures are silently dropped, no LOADcnt change.
    __builtin_prefetch(pA0 + 8, 0, 1);
    __builtin_prefetch(pA1 + 8, 0, 1);
    FragU a[2], b[4];
    a[0].u4[0] = pA0[0]; a[0].u4[1] = pA0[2];
    a[1].u4[0] = pA1[0]; a[1].u4[1] = pA1[2];
    b[0].u4[0] = pB0[0]; b[0].u4[1] = pB0[2];
    b[1].u4[0] = pB1[0]; b[1].u4[1] = pB1[2];
    b[2].u4[0] = pB2[0]; b[2].u4[1] = pB2[2];
    b[3].u4[0] = pB3[0]; b[3].u4[1] = pB3[2];
    pA0 += 4; pA1 += 4; pB0 += 4; pB1 += 4; pB2 += 4; pB3 += 4;
#pragma unroll
    for (int t = 0; t < 2; ++t)
#pragma unroll
      for (int u = 0; u < 4; ++u)
        acc[t][u] = __builtin_amdgcn_wmma_f32_16x16x32_bf16(
            false, a[t].v, false, b[u].v, (short)0, acc[t][u], false, false);
  }

  // epilogue: C/D layout VGPR j -> M = j + 8*kh, N = lane&15
#pragma unroll
  for (int u = 0; u < 4; ++u) {
    const int n = nBase + u * 16 + lm;
    const float bv = bias[n];
#pragma unroll
    for (int t = 0; t < 2; ++t) {
#pragma unroll
      for (int j = 0; j < 8; ++j) {
        const int m = mBase + t * 16 + j + kh * 8;
        Y[(size_t)m * N + n] = f2bf(fmaxf(acc[t][u][j] + bv, 0.f));
      }
    }
  }
}

// ---------------- final layer: sigmoid(X3 @ W3^T + b3), f32 out ----------------
__global__ void __launch_bounds__(256)
final_layer(const unsigned short* __restrict__ X,   // [NPTS,256] bf16
            const float* __restrict__ W3,           // [3,256]
            const float* __restrict__ b3,           // [3]
            float* __restrict__ out) {              // [NPTS,3]
  __shared__ float sW[3 * 256];
  for (int j = threadIdx.x; j < 3 * 256; j += 256) sW[j] = W3[j];
  __syncthreads();
  const int i = blockIdx.x * 256 + threadIdx.x;
  const unsigned short* xr = X + (size_t)i * 256;
  float a0 = 0.f, a1 = 0.f, a2 = 0.f;
  for (int k = 0; k < 256; ++k) {
    float xv = bf2f(xr[k]);
    a0 = fmaf(xv, sW[k], a0);
    a1 = fmaf(xv, sW[256 + k], a1);
    a2 = fmaf(xv, sW[512 + k], a2);
  }
  a0 += b3[0]; a1 += b3[1]; a2 += b3[2];
  out[i * 3 + 0] = 1.f / (1.f + __expf(-a0));
  out[i * 3 + 1] = 1.f / (1.f + __expf(-a1));
  out[i * 3 + 2] = 1.f / (1.f + __expf(-a2));
}

// ---------------- launcher ----------------
extern "C" void kernel_launch(void* const* d_in, const int* in_sizes, int n_in,
                              void* d_out, int out_size, void* d_ws, size_t ws_size,
                              hipStream_t stream) {
  const float* points    = (const float*)d_in[0];
  const float* normals   = (const float*)d_in[1];
  /* d_in[2] view_dirs: unused by reference */
  const float* feats     = (const float*)d_in[3];
  /* d_in[4] indices: unused */
  const float* particles = (const float*)d_in[5];
  const float* rays      = (const float*)d_in[6];
  const float* ro        = (const float*)d_in[7];
  const float* W0 = (const float*)d_in[8];   const float* b0 = (const float*)d_in[9];
  const float* W1 = (const float*)d_in[10];  const float* b1 = (const float*)d_in[11];
  const float* W2 = (const float*)d_in[12];  const float* b2 = (const float*)d_in[13];
  const float* W3 = (const float*)d_in[14];  const float* b3 = (const float*)d_in[15];

  char* ws = (char*)d_ws;
  const size_t szX0 = (size_t)NPTS * D_PAD * 2;   // 25165824
  const size_t szX  = (size_t)NPTS * D_HID * 2;   // 12582912
  unsigned short* X0  = (unsigned short*)(ws);
  unsigned short* X1  = (unsigned short*)(ws + szX0);
  unsigned short* X2  = (unsigned short*)(ws + szX0 + szX);
  unsigned short* X3  = X1;                        // alias: X1 dead after layer 2
  unsigned short* Wb0 = (unsigned short*)(ws + szX0 + 2 * szX);
  unsigned short* Wb1 = Wb0 + 256 * 512;
  unsigned short* Wb2 = Wb1 + 256 * 256;

  convert_weights<<<1024, 256, 0, stream>>>(W0, W1, W2, Wb0, Wb1, Wb2);
  build_features<<<NPTS / 256, 256, 0, stream>>>(points, normals, feats, particles,
                                                 rays, ro, X0);
  gemm_bf16_relu<<<NPTS / 64, 256, 0, stream>>>(X0, Wb0, b0, X1, 512, 256);
  gemm_bf16_relu<<<NPTS / 64, 256, 0, stream>>>(X1, Wb1, b1, X2, 256, 256);
  gemm_bf16_relu<<<NPTS / 64, 256, 0, stream>>>(X2, Wb2, b2, X3, 256, 256);
  final_layer<<<NPTS / 256, 256, 0, stream>>>(X3, W3, b3, (float*)d_out);
}
